// GNNBranch_89859305767780
// MI455X (gfx1250) — matrix operand order repaired and checked by reading, not compiled, not run on hardware
//
#include <hip/hip_runtime.h>
#include <hip/hip_bf16.h>
#include <math.h>
#include <stdint.h>

#define HID   128
#define HEADS 4
#define HDIM  32
#define QKVW  384   // q|k|v concatenated columns

typedef __bf16 v16bf __attribute__((ext_vector_type(16)));
typedef __bf16 v8bf  __attribute__((ext_vector_type(8)));
typedef float  v8f   __attribute__((ext_vector_type(8)));
typedef unsigned int u32x4 __attribute__((ext_vector_type(4)));
typedef int          i32x8 __attribute__((ext_vector_type(8)));
typedef int          i32x4 __attribute__((ext_vector_type(4)));

// ---------------------------------------------------------------------------
// helpers
// ---------------------------------------------------------------------------
__device__ __forceinline__ void atomicMaxFloat(float* addr, float val) {
  // monotonic int-bits trick (safe with -inf init; signed-max / unsigned-min
  // mix is consistent because positive-float bits < negative-float bits
  // in unsigned order).
  int iv = __float_as_int(val);
  if (iv >= 0) atomicMax((int*)addr, iv);
  else         atomicMin((unsigned int*)addr, (unsigned int)iv);
}

__device__ __forceinline__ float sigmoidf_fast(float x) {
  return 1.0f / (1.0f + __expf(-x));
}

// Stage a rows x HID bf16 matrix (row-major) from global into LDS.
// Preferred path: Tensor Data Mover (one wave issues a single 2-D descriptor,
// waits on TENSORcnt). Fallback: cooperative vector copy.
__device__ __forceinline__ void stage_weights_lds(const __bf16* __restrict__ W,
                                                  __bf16* lds, int rows) {
#if __has_builtin(__builtin_amdgcn_tensor_load_to_lds)
  if ((threadIdx.x >> 5) == 0) {   // wave 0 only; TDM ignores EXEC
    const uint64_t ga = (uint64_t)(uintptr_t)W;
    const uint32_t la = (uint32_t)(uintptr_t)(void*)lds;  // low 32 = LDS offs
    // ---- D# group 0: count=1 | lds_addr | global_addr[56:0] | type=2 ----
    u32x4 g0 = { 1u, la, (uint32_t)ga,
                 (uint32_t)((ga >> 32) & 0x01FFFFFFu) | (2u << 30) };
    // ---- D# group 1 (bit layout per CDNA5 ISA) ----
    i32x8 g1;
    g1[0] = (1 << 16);                          // wg_mask=0, data_size=1 (2B)
    g1[1] = (HID & 0xFFFF) << 16;               // tensor_dim0[15:0]  @ bits63:48
    g1[2] = (int)(((unsigned)rows & 0xFFFF) << 16); // dim0 hi=0 | tensor_dim1 lo
    g1[3] = (HID << 16);                        // dim1 hi=0 | tile_dim0=HID
    g1[4] = rows;                               // tile_dim1=rows, tile_dim2=0
    g1[5] = HID;                                // tensor_dim0_stride lo32
    g1[6] = 0;                                  // stride hi | dim1_stride lo
    g1[7] = 0;
    i32x4 g2 = { 0, 0, 0, 0 };
    i32x4 g3 = { 0, 0, 0, 0 };
#if __clang_major__ >= 23
    i32x8 g4 = { 0, 0, 0, 0, 0, 0, 0, 0 };
    __builtin_amdgcn_tensor_load_to_lds(g0, g1, g2, g3, g4, 0);
#else
    __builtin_amdgcn_tensor_load_to_lds(g0, g1, g2, g3, 0);
#endif
    __builtin_amdgcn_s_wait_tensorcnt(0);
  }
  __syncthreads();
#else
  const int nvec = rows * HID / 8;              // 16B chunks
  for (int i = threadIdx.x; i < nvec; i += (int)blockDim.x)
    *(v8bf*)&lds[i * 8] = *(const v8bf*)&W[i * 8];
  __syncthreads();
#endif
}

// ---------------------------------------------------------------------------
// 0) convert weights to bf16: Wqkv (384x128, rows = output cols) and Wo
// ---------------------------------------------------------------------------
__global__ void gnn_cast_weights(const float* __restrict__ Wq,
                                 const float* __restrict__ Wk,
                                 const float* __restrict__ Wv,
                                 const float* __restrict__ Wo,
                                 __bf16* __restrict__ Wqkv_bf,
                                 __bf16* __restrict__ Wo_bf) {
  int i = blockIdx.x * blockDim.x + threadIdx.x;
  const int per = HID * HID; // 16384
  if (i < 3 * per) {
    float v = (i < per) ? Wq[i] : (i < 2 * per) ? Wk[i - per] : Wv[i - 2 * per];
    Wqkv_bf[i] = (__bf16)v;
  }
  if (i < per) Wo_bf[i] = (__bf16)Wo[i];
}

// ---------------------------------------------------------------------------
// 1) h_combined = h + frac_coords @ We^T + be  -> bf16
// ---------------------------------------------------------------------------
__global__ void gnn_embed_cast(const float* __restrict__ h,
                               const float* __restrict__ frac,
                               const float* __restrict__ We,
                               const float* __restrict__ be,
                               __bf16* __restrict__ hc_bf, int n) {
  int i = blockIdx.x * blockDim.x + threadIdx.x;
  if (i >= n * HID) return;
  int r = i >> 7, c = i & 127;
  float v = h[i] + be[c];
  v = fmaf(frac[r * 3 + 0], We[c * 3 + 0], v);
  v = fmaf(frac[r * 3 + 1], We[c * 3 + 1], v);
  v = fmaf(frac[r * 3 + 2], We[c * 3 + 2], v);
  hc_bf[i] = (__bf16)v;
}

// ---------------------------------------------------------------------------
// 2) init: agg = 0, smax = -inf, ssum = 0
// ---------------------------------------------------------------------------
__global__ void gnn_init(float* __restrict__ agg, float* __restrict__ smax,
                         float* __restrict__ ssum, int n) {
  int i = blockIdx.x * blockDim.x + threadIdx.x;
  if (i < n * HID) agg[i] = 0.0f;
  if (i < n * HEADS) { smax[i] = -__builtin_inff(); ssum[i] = 0.0f; }
}

// ---------------------------------------------------------------------------
// 3) QKV GEMM: qkv[n,384] = hc[n,128] @ Wqkv^T + bias   (bf16 WMMA, f32 acc)
//    256 threads = 8 waves; block owns 128 rows (wave w -> 16-row tile w);
//    full Wqkv staged once per block into LDS via the Tensor Data Mover.
// ---------------------------------------------------------------------------
__global__ void __launch_bounds__(256)
gnn_qkv_wmma(const __bf16* __restrict__ A, const __bf16* __restrict__ W,
             const float* __restrict__ bq, const float* __restrict__ bk,
             const float* __restrict__ bv, __bf16* __restrict__ qkv, int n) {
  __shared__ __align__(128) __bf16 ldsW[QKVW * HID];   // 96 KB of 320 KB WGP LDS

  stage_weights_lds(W, ldsW, QKVW);

  const int wave = threadIdx.x >> 5;
  const int lane = threadIdx.x & 31;
  const int sub  = lane & 15;
  const bool hiHalf = lane >= 16;
  const int rowBase = blockIdx.x * 128 + wave * 16;

  // ---- A fragments (16 rows x K=128), reused for all 24 column tiles ----
  const int ar = rowBase + sub;
  const bool arValid = ar < n;
  const __bf16* arow = A + (size_t)(arValid ? ar : 0) * HID;
  v16bf afrag[4];
#pragma unroll
  for (int ks = 0; ks < 4; ++ks) {
    const int kb = ks * 32 + (hiHalf ? 16 : 0);
    v8bf lo = {}, hi = {};
    if (arValid) {
      lo = *(const v8bf*)(arow + kb);
      hi = *(const v8bf*)(arow + kb + 8);
    }
    afrag[ks] = __builtin_shufflevector(lo, hi, 0, 1, 2, 3, 4, 5, 6, 7,
                                        8, 9, 10, 11, 12, 13, 14, 15);
  }

  // ---- every wave sweeps all 24 q|k|v column tiles; B frags from LDS ----
  for (int t = 0; t < QKVW / 16; ++t) {
    const int bcol = t * 16 + sub;
    const __bf16* wrow = ldsW + (size_t)bcol * HID + (hiHalf ? 16 : 0);
    v8f acc = {};
#pragma unroll
    for (int ks = 0; ks < 4; ++ks) {
      v16bf bfrag = *(const v16bf*)(wrow + ks * 32);
      acc = __builtin_amdgcn_wmma_f32_16x16x32_bf16(
          false, afrag[ks], false, bfrag, (short)0, acc, false, false);
    }
    const float bias = (bcol < HID) ? bq[bcol]
                     : (bcol < 2 * HID) ? bk[bcol - HID] : bv[bcol - 2 * HID];
#pragma unroll
    for (int r = 0; r < 8; ++r) {
      const int row = rowBase + r + (hiHalf ? 8 : 0);
      if (row < n) qkv[(size_t)row * QKVW + bcol] = (__bf16)(acc[r] + bias);
    }
  }
}

// ---------------------------------------------------------------------------
// 4) per-edge gate + attention scores + segment max (atomic)
// ---------------------------------------------------------------------------
__global__ void gnn_edge_score(const __bf16* __restrict__ qkv,
                               const int* __restrict__ ei,
                               const float* __restrict__ dist,
                               const float* __restrict__ Wg1,
                               const float* __restrict__ bg1,
                               const float* __restrict__ Wg2,
                               const float* __restrict__ bg2,
                               float* __restrict__ scores,
                               float* __restrict__ smax, int eTot) {
  int e = blockIdx.x * blockDim.x + threadIdx.x;
  if (e >= eTot) return;
  const int row = ei[e];
  const int col = ei[eTot + e];

  // prefetch a future edge's gather targets (lowers to global_prefetch_b8)
  if (e + 2048 < eTot) {
    __builtin_prefetch(qkv + (size_t)ei[e + 2048] * QKVW, 0, 1);
    __builtin_prefetch(qkv + (size_t)ei[eTot + e + 2048] * QKVW + HID, 0, 1);
  }

  // edge gate MLP: 1 -> 32 (SiLU) -> 1 (sigmoid)
  const float d = dist[e];
  float acc = bg2[0];
#pragma unroll 8
  for (int j = 0; j < 32; ++j) {
    float t = fmaf(d, Wg1[j], bg1[j]);
    acc = fmaf(Wg2[j], t * sigmoidf_fast(t), acc);
  }
  const float g = sigmoidf_fast(acc);

  const __bf16* qrow = qkv + (size_t)row * QKVW;           // q at dest
  const __bf16* krow = qkv + (size_t)col * QKVW + HID;     // k at source
  const float scale = 0.1767766952966369f * g;             // g / sqrt(32)
#pragma unroll
  for (int hh = 0; hh < HEADS; ++hh) {
    float s = 0.0f;
#pragma unroll 8
    for (int dd = 0; dd < HDIM; ++dd)
      s = fmaf((float)qrow[hh * HDIM + dd], (float)krow[hh * HDIM + dd], s);
    const float sc = s * scale;
    scores[(size_t)e * HEADS + hh] = sc;
    atomicMaxFloat(&smax[row * HEADS + hh], sc);
  }
}

// ---------------------------------------------------------------------------
// 5) ex = exp(score - smax[row]);  ssum += ex  (per edge*head)
// ---------------------------------------------------------------------------
__global__ void gnn_edge_exp(const int* __restrict__ ei,
                             float* __restrict__ scores,
                             const float* __restrict__ smax,
                             float* __restrict__ ssum, int eTot) {
  int i = blockIdx.x * blockDim.x + threadIdx.x;
  if (i >= eTot * HEADS) return;
  const int e = i >> 2, hh = i & 3;
  const int row = ei[e];
  const float ex = __expf(scores[i] - smax[row * HEADS + hh]);
  scores[i] = ex;
  atomicAdd(&ssum[row * HEADS + hh], ex);
}

// ---------------------------------------------------------------------------
// 6) wave-per-edge: agg[row] += (ex/ssum) * v[col]   (128 f32 atomics / edge)
// ---------------------------------------------------------------------------
__global__ void gnn_edge_agg(const __bf16* __restrict__ qkv,
                             const int* __restrict__ ei,
                             const float* __restrict__ scores,
                             const float* __restrict__ ssum,
                             float* __restrict__ agg, int eTot) {
  const long long gid = (long long)blockIdx.x * blockDim.x + threadIdx.x;
  const long long e = gid >> 5;
  const int lane = (int)(gid & 31);
  if (e >= eTot) return;
  const int row = ei[e];
  const int col = ei[eTot + e];
  const __bf16* vrow = qkv + (size_t)col * QKVW + 2 * HID;
  float* arow = agg + (size_t)row * HID;
#pragma unroll
  for (int hh = 0; hh < HEADS; ++hh) {
    const float attn = scores[(size_t)e * HEADS + hh] /
                       (ssum[row * HEADS + hh] + 1e-16f);
    atomicAdd(&arow[hh * HDIM + lane], attn * (float)vrow[hh * HDIM + lane]);
  }
}

// ---------------------------------------------------------------------------
// 7) agg f32 -> bf16
// ---------------------------------------------------------------------------
__global__ void gnn_cast_agg(const float* __restrict__ agg,
                             __bf16* __restrict__ agg_bf, int n) {
  int i = blockIdx.x * blockDim.x + threadIdx.x;
  if (i < n * HID) agg_bf[i] = (__bf16)agg[i];
}

// ---------------------------------------------------------------------------
// 8) out GEMM: out[n,128] = agg[n,128] @ Wo^T + bo   (bf16 WMMA, f32 out)
//    same blocking as QKV GEMM; Wo (32 KB) staged via TDM.
// ---------------------------------------------------------------------------
__global__ void __launch_bounds__(256)
gnn_out_wmma(const __bf16* __restrict__ A, const __bf16* __restrict__ W,
             const float* __restrict__ bo, float* __restrict__ out, int n) {
  __shared__ __align__(128) __bf16 ldsW[HID * HID];    // 32 KB

  stage_weights_lds(W, ldsW, HID);

  const int wave = threadIdx.x >> 5;
  const int lane = threadIdx.x & 31;
  const int sub  = lane & 15;
  const bool hiHalf = lane >= 16;
  const int rowBase = blockIdx.x * 128 + wave * 16;

  const int ar = rowBase + sub;
  const bool arValid = ar < n;
  const __bf16* arow = A + (size_t)(arValid ? ar : 0) * HID;
  v16bf afrag[4];
#pragma unroll
  for (int ks = 0; ks < 4; ++ks) {
    const int kb = ks * 32 + (hiHalf ? 16 : 0);
    v8bf lo = {}, hi = {};
    if (arValid) {
      lo = *(const v8bf*)(arow + kb);
      hi = *(const v8bf*)(arow + kb + 8);
    }
    afrag[ks] = __builtin_shufflevector(lo, hi, 0, 1, 2, 3, 4, 5, 6, 7,
                                        8, 9, 10, 11, 12, 13, 14, 15);
  }

  for (int t = 0; t < HID / 16; ++t) {     // 8 column tiles per wave
    const int bcol = t * 16 + sub;
    const __bf16* wrow = ldsW + (size_t)bcol * HID + (hiHalf ? 16 : 0);
    v8f acc = {};
#pragma unroll
    for (int ks = 0; ks < 4; ++ks) {
      v16bf bfrag = *(const v16bf*)(wrow + ks * 32);
      acc = __builtin_amdgcn_wmma_f32_16x16x32_bf16(
          false, afrag[ks], false, bfrag, (short)0, acc, false, false);
    }
    const float bias = bo[bcol];
#pragma unroll
    for (int r = 0; r < 8; ++r) {
      const int row = rowBase + r + (hiHalf ? 8 : 0);
      if (row < n) out[(size_t)row * HID + bcol] = acc[r] + bias;
    }
  }
}

// ---------------------------------------------------------------------------
// host-side launcher
// ---------------------------------------------------------------------------
extern "C" void kernel_launch(void* const* d_in, const int* in_sizes, int n_in,
                              void* d_out, int out_size, void* d_ws, size_t ws_size,
                              hipStream_t stream) {
  const float* h     = (const float*)d_in[0];
  const float* frac  = (const float*)d_in[1];
  const float* dist  = (const float*)d_in[2];
  const int*   ei    = (const int*)d_in[3];
  const float* We    = (const float*)d_in[4];
  const float* be    = (const float*)d_in[5];
  const float* Wq    = (const float*)d_in[6];
  const float* bq    = (const float*)d_in[7];
  const float* Wk    = (const float*)d_in[8];
  const float* bk    = (const float*)d_in[9];
  const float* Wv    = (const float*)d_in[10];
  const float* bv    = (const float*)d_in[11];
  const float* Wo    = (const float*)d_in[12];
  const float* bo    = (const float*)d_in[13];
  const float* Wg1   = (const float*)d_in[14];
  const float* bg1   = (const float*)d_in[15];
  const float* Wg2   = (const float*)d_in[16];
  const float* bg2   = (const float*)d_in[17];
  float* out = (float*)d_out;

  const int n = in_sizes[0] / HID;   // 50000
  const int e = in_sizes[2];         // 800000

  // ---- workspace carve-up (256B aligned) ----
  char* ws = (char*)d_ws;
  size_t off = 0;
  auto carve = [&](size_t bytes) -> char* {
    char* p = ws + off;
    off = (off + bytes + 255) & ~(size_t)255;
    return p;
  };
  __bf16* hc_bf   = (__bf16*)carve((size_t)n * HID * 2);
  __bf16* Wqkv_bf = (__bf16*)carve((size_t)QKVW * HID * 2);
  __bf16* Wo_bf   = (__bf16*)carve((size_t)HID * HID * 2);
  __bf16* qkv_bf  = (__bf16*)carve((size_t)n * QKVW * 2);
  float*  scores  = (float*)carve((size_t)e * HEADS * 4);
  float*  smax    = (float*)carve((size_t)n * HEADS * 4);
  float*  ssum    = (float*)carve((size_t)n * HEADS * 4);
  float*  agg     = (float*)carve((size_t)n * HID * 4);
  __bf16* agg_bf  = (__bf16*)carve((size_t)n * HID * 2);
  (void)ws_size; (void)n_in; (void)out_size;

  const int T = 256;
  // 0) weights -> bf16
  gnn_cast_weights<<<(3 * HID * HID + T - 1) / T, T, 0, stream>>>(
      Wq, Wk, Wv, Wo, Wqkv_bf, Wo_bf);
  // 1) h_combined -> bf16
  gnn_embed_cast<<<((size_t)n * HID + T - 1) / T, T, 0, stream>>>(
      h, frac, We, be, hc_bf, n);
  // 2) init agg / smax / ssum
  gnn_init<<<((size_t)n * HID + T - 1) / T, T, 0, stream>>>(agg, smax, ssum, n);
  // 3) QKV projection (WMMA, 128 rows/block, TDM-staged weights)
  gnn_qkv_wmma<<<(n + 127) / 128, 256, 0, stream>>>(
      hc_bf, Wqkv_bf, bq, bk, bv, qkv_bf, n);
  // 4) edge gate + scores + segment max
  gnn_edge_score<<<(e + T - 1) / T, T, 0, stream>>>(
      qkv_bf, ei, dist, Wg1, bg1, Wg2, bg2, scores, smax, e);
  // 5) exp + segment sum
  gnn_edge_exp<<<((size_t)e * HEADS + T - 1) / T, T, 0, stream>>>(
      ei, scores, smax, ssum, e);
  // 6) weighted scatter-add aggregation (wave per edge)
  gnn_edge_agg<<<((size_t)e * 32 + T - 1) / T, T, 0, stream>>>(
      qkv_bf, ei, scores, ssum, agg, e);
  // 7) agg -> bf16
  gnn_cast_agg<<<((size_t)n * HID + T - 1) / T, T, 0, stream>>>(agg, agg_bf, n);
  // 8) output projection (WMMA, TDM-staged Wo)
  gnn_out_wmma<<<(n + 127) / 128, 256, 0, stream>>>(agg_bf, Wo_bf, bo, out, n);
}
